// Net_57071525429594
// MI455X (gfx1250) — matrix operand order repaired and checked by reading, not compiled, not run on hardware
//
#include <hip/hip_runtime.h>
#include <hip/hip_bf16.h>
#include <math.h>

typedef __bf16 bf16_t;
typedef __attribute__((ext_vector_type(16))) __bf16 v16bf;
typedef __attribute__((ext_vector_type(8)))  float  v8f;

#define N_NODES     25000
#define N_PAD       25008           // 16-row padded
#define N_EDGES     400000
#define FEAT        128
#define HID         64
#define HEADS       4
#define NUM_CLASSES 2
#define NUM_GRAPHS  64
#define GAT_DIM     (HEADS * HID)   // 256

static __device__ __forceinline__ float lrelu(float x) { return x > 0.f ? x : 0.2f * x; }

union FragBF {
    v16bf v;
    uint4 q[2];
};

// ---------------------------------------------------------------- utility ---
__global__ void k_fill_u32(unsigned int* __restrict__ p, unsigned int v, int n) {
    int i = blockIdx.x * blockDim.x + threadIdx.x;
    if (i < n) p[i] = v;
}

__global__ void k_f32_to_bf16(const float* __restrict__ in, bf16_t* __restrict__ out, int n) {
    int i = blockIdx.x * blockDim.x + threadIdx.x;
    if (i < n) out[i] = (bf16_t)in[i];
}

// Pack f32 weight [K,N] into per-lane WMMA B-fragment layout:
// Wout[((ks*ntiles + nt)*32 + lane)*16 + e] = W[(ks*32 + 8*(lane>>4) + e + 8*(e>>3))*N + nt*16 + (lane&15)]
__global__ void k_pack_w(const float* __restrict__ Win, bf16_t* __restrict__ Wout, int K, int N) {
    int idx = blockIdx.x * blockDim.x + threadIdx.x;
    if (idx >= K * N) return;
    int e      = idx & 15;
    int lane   = (idx >> 4) & 31;
    int tile   = idx >> 9;
    int ntiles = N >> 4;
    int nt     = tile % ntiles;
    int ks     = tile / ntiles;
    int k      = ks * 32 + ((lane >> 4) << 3) + e + ((e >> 3) << 3);
    int n      = nt * 16 + (lane & 15);
    Wout[idx] = (bf16_t)Win[(size_t)k * N + n];
}

// ------------------------------------------------------------ edge scatter ---
template <int C>
__global__ void k_scatter_add_f32(const float* __restrict__ feat, const int* __restrict__ src,
                                  const int* __restrict__ dst, float* __restrict__ agg) {
    long long i = (long long)blockIdx.x * blockDim.x + threadIdx.x;
    const long long total = (long long)N_EDGES * (C / 4);
    if (i >= total) return;
    int e  = (int)(i / (C / 4));
    int c4 = ((int)(i % (C / 4))) * 4;
    const float4 v = *(const float4*)(feat + (size_t)src[e] * C + c4);
    float* base = agg + (size_t)dst[e] * C + c4;
    atomicAdd(base + 0, v.x);
    atomicAdd(base + 1, v.y);
    atomicAdd(base + 2, v.z);
    atomicAdd(base + 3, v.w);
}

template <int C>
__global__ void k_scatter_add_bf16(const bf16_t* __restrict__ feat, const int* __restrict__ src,
                                   const int* __restrict__ dst, float* __restrict__ agg) {
    long long i = (long long)blockIdx.x * blockDim.x + threadIdx.x;
    const long long total = (long long)N_EDGES * (C / 4);
    if (i >= total) return;
    int e  = (int)(i / (C / 4));
    int c4 = ((int)(i % (C / 4))) * 4;
    union { uint2 u; bf16_t h[4]; } raw;
    raw.u = *(const uint2*)(feat + (size_t)src[e] * C + c4);
    float* base = agg + (size_t)dst[e] * C + c4;
    atomicAdd(base + 0, (float)raw.h[0]);
    atomicAdd(base + 1, (float)raw.h[1]);
    atomicAdd(base + 2, (float)raw.h[2]);
    atomicAdd(base + 3, (float)raw.h[3]);
}

// --------------------------------------------------------------- WMMA GEMM ---
// Out[M,N] = act( A1[M,K](f32) @ W1 + A2[M,K](bf16) @ W2 + bias ), W pre-packed.
// One wave -> one 16x16 D tile; v_wmma_f32_16x16x32_bf16 over K in steps of 32.
// A-frag per lane: two contiguous 8-elem runs -> 2x b128 loads.
// B-frag per lane: pre-packed contiguous 16 elems -> 2x b128 loads.
template <bool HAS_A1, bool RELU, bool HAS_BIAS, bool OUTF, bool OUTB>
__global__ void __launch_bounds__(128)
k_wmma_gemm(const float* __restrict__ A1, const bf16_t* __restrict__ A2,
            const bf16_t* __restrict__ W1p, const bf16_t* __restrict__ W2p,
            const float* __restrict__ bias,
            float* __restrict__ Outf, bf16_t* __restrict__ Outb,
            int M, int K, int N) {
    const int wave = threadIdx.x >> 5;
    const int lane = threadIdx.x & 31;
    const int half = lane >> 4;
    const int r    = lane & 15;
    const int M0   = blockIdx.x * 16;

    int rowA = M0 + r;
    if (rowA > M - 1) rowA = M - 1;           // clamp: keeps EXEC all-ones around WMMA
    const size_t aoff = (size_t)rowA * K;
    const int ntiles = N >> 4;

    for (int nt = wave; nt < ntiles; nt += 4) {
        const int ncol = nt * 16 + r;
        v8f acc = {};
        for (int k0 = 0; k0 < K; k0 += 32) {
            const size_t wtile = ((size_t)(k0 >> 5) * ntiles + nt) * 512 + (lane << 4);

            FragBF a2, b2;
            a2.q[0] = *(const uint4*)(A2 + aoff + k0 + half * 8);
            a2.q[1] = *(const uint4*)(A2 + aoff + k0 + 16 + half * 8);
            b2.q[0] = *(const uint4*)(W2p + wtile);
            b2.q[1] = *(const uint4*)(W2p + wtile + 8);
            acc = __builtin_amdgcn_wmma_f32_16x16x32_bf16(false, a2.v, false, b2.v,
                                                          (short)0, acc, false, false);
            if (HAS_A1) {
                const float* arow = A1 + aoff + k0 + half * 8;
                float4 f0 = *(const float4*)(arow);
                float4 f1 = *(const float4*)(arow + 4);
                float4 f2 = *(const float4*)(arow + 16);
                float4 f3 = *(const float4*)(arow + 20);
                float fa[16] = { f0.x, f0.y, f0.z, f0.w, f1.x, f1.y, f1.z, f1.w,
                                 f2.x, f2.y, f2.z, f2.w, f3.x, f3.y, f3.z, f3.w };
                FragBF a1, b1;
#pragma unroll
                for (int e = 0; e < 16; ++e) a1.v[e] = (bf16_t)fa[e];
                b1.q[0] = *(const uint4*)(W1p + wtile);
                b1.q[1] = *(const uint4*)(W1p + wtile + 8);
                acc = __builtin_amdgcn_wmma_f32_16x16x32_bf16(false, a1.v, false, b1.v,
                                                              (short)0, acc, false, false);
            }
        }
        const float bval = HAS_BIAS ? bias[ncol] : 0.f;
#pragma unroll
        for (int j = 0; j < 8; ++j) {
            const int orow = M0 + j + half * 8;           // outputs padded to N_PAD rows
            float v = acc[j] + bval;
            if (RELU) v = v > 0.f ? v : 0.f;
            if (OUTF) Outf[(size_t)orow * N + ncol] = v;
            if (OUTB) Outb[(size_t)orow * N + ncol] = (bf16_t)v;
        }
    }
}

// --------------------------------------------------------------------- GAT ---
__global__ void k_attn_coef(const float* __restrict__ xl, const float* __restrict__ att_src,
                            const float* __restrict__ att_dst,
                            float* __restrict__ as_, float* __restrict__ ad_) {
    int i = blockIdx.x * blockDim.x + threadIdx.x;
    if (i >= N_NODES * HEADS) return;
    int n = i / HEADS, h = i % HEADS;
    const float* xrow = xl + (size_t)n * GAT_DIM + h * HID;
    float s = 0.f, d = 0.f;
    for (int c = 0; c < HID; ++c) {
        float v = xrow[c];
        s += v * att_src[h * HID + c];
        d += v * att_dst[h * HID + c];
    }
    as_[i] = s;
    ad_[i] = d;
}

static __device__ __forceinline__ void atomicMaxF(float* addr, float val) {
    if (val >= 0.f) atomicMax((int*)addr, __float_as_int(val));
    else            atomicMin((unsigned int*)addr, __float_as_uint(val));
}

__global__ void k_edge_max(const int* __restrict__ src, const int* __restrict__ dst,
                           const float* __restrict__ as_, const float* __restrict__ ad_,
                           float* __restrict__ m_) {
    int i = blockIdx.x * blockDim.x + threadIdx.x;
    const int total = (N_EDGES + N_NODES) * HEADS;   // edges + self-loops
    if (i >= total) return;
    int t = i / HEADS, h = i % HEADS;
    int s, d;
    if (t < N_EDGES) { s = src[t]; d = dst[t]; } else { s = d = t - N_EDGES; }
    float e = lrelu(as_[s * HEADS + h] + ad_[d * HEADS + h]);
    atomicMaxF(&m_[d * HEADS + h], e);
}

__global__ void k_edge_denom(const int* __restrict__ src, const int* __restrict__ dst,
                             const float* __restrict__ as_, const float* __restrict__ ad_,
                             const float* __restrict__ m_, float* __restrict__ den_) {
    int i = blockIdx.x * blockDim.x + threadIdx.x;
    const int total = (N_EDGES + N_NODES) * HEADS;
    if (i >= total) return;
    int t = i / HEADS, h = i % HEADS;
    int s, d;
    if (t < N_EDGES) { s = src[t]; d = dst[t]; } else { s = d = t - N_EDGES; }
    float e = lrelu(as_[s * HEADS + h] + ad_[d * HEADS + h]);
    atomicAdd(&den_[d * HEADS + h], __expf(e - m_[d * HEADS + h]));
}

// one thread = (edge, head, 16-channel quarter? no: 4-channel group) -> p computed once per 4 chans
__global__ void k_edge_aggregate(const int* __restrict__ src, const int* __restrict__ dst,
                                 const float* __restrict__ as_, const float* __restrict__ ad_,
                                 const float* __restrict__ m_, const float* __restrict__ xl,
                                 float* __restrict__ oacc) {
    long long i = (long long)blockIdx.x * blockDim.x + threadIdx.x;
    const long long total = (long long)(N_EDGES + N_NODES) * (GAT_DIM / 4);
    if (i >= total) return;
    int t  = (int)(i / (GAT_DIM / 4));
    int c4 = ((int)(i % (GAT_DIM / 4))) * 4;
    int h  = c4 / HID;
    int s, d;
    if (t < N_EDGES) { s = src[t]; d = dst[t]; } else { s = d = t - N_EDGES; }
    float e = lrelu(as_[s * HEADS + h] + ad_[d * HEADS + h]);
    float p = __expf(e - m_[d * HEADS + h]);
    const float4 v = *(const float4*)(xl + (size_t)s * GAT_DIM + c4);
    float* base = oacc + (size_t)d * GAT_DIM + c4;
    atomicAdd(base + 0, p * v.x);
    atomicAdd(base + 1, p * v.y);
    atomicAdd(base + 2, p * v.z);
    atomicAdd(base + 3, p * v.w);
}

__global__ void k_gat_finalize(const float* __restrict__ oacc, const float* __restrict__ den_,
                               const float* __restrict__ bg, bf16_t* __restrict__ h1) {
    int i = blockIdx.x * blockDim.x + threadIdx.x;
    if (i >= N_NODES * GAT_DIM) return;
    int n = i / GAT_DIM, c = i % GAT_DIM, h = c / HID;
    float v = oacc[i] / den_[n * HEADS + h] + bg[c];
    v = v > 0.f ? v : 0.f;
    h1[i] = (bf16_t)v;
}

// ------------------------------------------------------------- pool + head ---
__global__ void k_pool(const float* __restrict__ h2, const int* __restrict__ batch,
                       float* __restrict__ g) {
    int i = blockIdx.x * blockDim.x + threadIdx.x;
    if (i >= N_NODES * HID) return;
    int n = i / HID, c = i % HID;
    atomicAdd(&g[batch[n] * HID + c], h2[i]);
}

__global__ void k_fc1(const float* __restrict__ g, const float* __restrict__ W,
                      const float* __restrict__ b, float* __restrict__ t1) {
    int gi = blockIdx.x, j = threadIdx.x;
    float s = b[j];
    for (int c = 0; c < HID; ++c) s += g[gi * HID + c] * W[c * HID + j];
    t1[gi * HID + j] = s > 0.f ? s : 0.f;
}

__global__ void k_fc2(const float* __restrict__ t1, const float* __restrict__ W,
                      const float* __restrict__ b, float* __restrict__ out) {
    int i = threadIdx.x;                       // 0 .. NUM_GRAPHS*NUM_CLASSES-1
    if (i >= NUM_GRAPHS * NUM_CLASSES) return;
    int gi = i / NUM_CLASSES, j = i % NUM_CLASSES;
    float s = b[j];
    for (int c = 0; c < HID; ++c) s += t1[gi * HID + c] * W[c * NUM_CLASSES + j];
    out[i] = 1.f / (1.f + __expf(-s));
}

// ------------------------------------------------------------------ driver ---
extern "C" void kernel_launch(void* const* d_in, const int* in_sizes, int n_in,
                              void* d_out, int out_size, void* d_ws, size_t ws_size,
                              hipStream_t stream) {
    const float* x       = (const float*)d_in[0];
    const int*   ei      = (const int*)d_in[1];
    const int*   batch   = (const int*)d_in[2];
    const float* W1_rel  = (const float*)d_in[3];
    const float* b1      = (const float*)d_in[4];
    const float* W1_root = (const float*)d_in[5];
    const float* Wg      = (const float*)d_in[6];
    const float* att_src = (const float*)d_in[7];
    const float* att_dst = (const float*)d_in[8];
    const float* bg      = (const float*)d_in[9];
    const float* W5_rel  = (const float*)d_in[10];
    const float* b5      = (const float*)d_in[11];
    const float* W5_root = (const float*)d_in[12];
    const float* W_fc1   = (const float*)d_in[13];
    const float* b_fc1   = (const float*)d_in[14];
    const float* W_fc2   = (const float*)d_in[15];
    const float* b_fc2   = (const float*)d_in[16];
    const int* src = ei;
    const int* dst = ei + N_EDGES;

    // ---- workspace arena (256B aligned slots) ----
    char*  ws  = (char*)d_ws;
    size_t off = 0;
    auto alloc = [&](size_t bytes) -> char* {
        char* p = ws + off;
        off += (bytes + 255) & ~(size_t)255;
        return p;
    };
    bf16_t* xbf  = (bf16_t*)alloc((size_t)N_NODES * FEAT * 2);
    bf16_t* wW1r = (bf16_t*)alloc((size_t)FEAT * HID * 2);       // packed fragments
    bf16_t* wW1o = (bf16_t*)alloc((size_t)FEAT * HID * 2);
    bf16_t* wWg  = (bf16_t*)alloc((size_t)HID * GAT_DIM * 2);
    bf16_t* wW5r = (bf16_t*)alloc((size_t)GAT_DIM * HID * 2);
    bf16_t* wW5o = (bf16_t*)alloc((size_t)GAT_DIM * HID * 2);
    float*  agg1 = (float*)alloc((size_t)N_PAD * FEAT * 4);      // later reused as h2 (N_PAD*HID)
    bf16_t* h0bf = (bf16_t*)alloc((size_t)N_PAD * HID * 2);
    float*  xl   = (float*)alloc((size_t)N_PAD * GAT_DIM * 4);   // later reused as agg5
    float*  as_  = (float*)alloc((size_t)N_NODES * HEADS * 4);
    float*  ad_  = (float*)alloc((size_t)N_NODES * HEADS * 4);
    float*  m_   = (float*)alloc((size_t)N_NODES * HEADS * 4);
    float*  den_ = (float*)alloc((size_t)N_NODES * HEADS * 4);
    float*  oacc = (float*)alloc((size_t)N_NODES * GAT_DIM * 4);
    bf16_t* h1bf = (bf16_t*)alloc((size_t)N_NODES * GAT_DIM * 2);
    float*  g    = (float*)alloc((size_t)NUM_GRAPHS * HID * 4);
    float*  t1   = (float*)alloc((size_t)NUM_GRAPHS * HID * 4);
    float*  h2   = agg1;   // safe: agg1 dead after conv1 GEMM
    float*  agg5 = xl;     // safe: xl dead after edge aggregation

    auto cdiv = [](long long a, long long b) { return (int)((a + b - 1) / b); };
    const int gblocks = N_PAD / 16;

    // ---- precision conversion + weight fragment packing ----
    k_f32_to_bf16<<<cdiv((long long)N_NODES * FEAT, 256), 256, 0, stream>>>(x, xbf, N_NODES * FEAT);
    k_pack_w<<<cdiv(FEAT * HID, 256), 256, 0, stream>>>(W1_rel, wW1r, FEAT, HID);
    k_pack_w<<<cdiv(FEAT * HID, 256), 256, 0, stream>>>(W1_root, wW1o, FEAT, HID);
    k_pack_w<<<cdiv(HID * GAT_DIM, 256), 256, 0, stream>>>(Wg, wWg, HID, GAT_DIM);
    k_pack_w<<<cdiv(GAT_DIM * HID, 256), 256, 0, stream>>>(W5_rel, wW5r, GAT_DIM, HID);
    k_pack_w<<<cdiv(GAT_DIM * HID, 256), 256, 0, stream>>>(W5_root, wW5o, GAT_DIM, HID);

    // ---- conv1: GraphConv(128 -> 64) ----
    k_fill_u32<<<cdiv((long long)N_NODES * FEAT, 256), 256, 0, stream>>>((unsigned*)agg1, 0u, N_NODES * FEAT);
    k_scatter_add_f32<FEAT><<<cdiv((long long)N_EDGES * (FEAT / 4), 256), 256, 0, stream>>>(x, src, dst, agg1);
    k_wmma_gemm<true, true, true, false, true><<<gblocks, 128, 0, stream>>>(
        agg1, xbf, wW1r, wW1o, b1, nullptr, h0bf, N_NODES, FEAT, HID);

    // ---- GAT(64 -> 4x64) ----
    k_wmma_gemm<false, false, false, true, false><<<gblocks, 128, 0, stream>>>(
        nullptr, h0bf, nullptr, wWg, nullptr, xl, nullptr, N_NODES, HID, GAT_DIM);
    k_attn_coef<<<cdiv(N_NODES * HEADS, 256), 256, 0, stream>>>(xl, att_src, att_dst, as_, ad_);
    k_fill_u32<<<cdiv(N_NODES * HEADS, 256), 256, 0, stream>>>((unsigned*)m_, 0xFF800000u, N_NODES * HEADS);
    k_fill_u32<<<cdiv(N_NODES * HEADS, 256), 256, 0, stream>>>((unsigned*)den_, 0u, N_NODES * HEADS);
    k_fill_u32<<<cdiv((long long)N_NODES * GAT_DIM, 256), 256, 0, stream>>>((unsigned*)oacc, 0u, N_NODES * GAT_DIM);
    k_edge_max<<<cdiv((long long)(N_EDGES + N_NODES) * HEADS, 256), 256, 0, stream>>>(src, dst, as_, ad_, m_);
    k_edge_denom<<<cdiv((long long)(N_EDGES + N_NODES) * HEADS, 256), 256, 0, stream>>>(src, dst, as_, ad_, m_, den_);
    k_edge_aggregate<<<cdiv((long long)(N_EDGES + N_NODES) * (GAT_DIM / 4), 256), 256, 0, stream>>>(
        src, dst, as_, ad_, m_, xl, oacc);
    k_gat_finalize<<<cdiv((long long)N_NODES * GAT_DIM, 256), 256, 0, stream>>>(oacc, den_, bg, h1bf);

    // ---- conv5: GraphConv(256 -> 64) ----
    k_fill_u32<<<cdiv((long long)N_NODES * GAT_DIM, 256), 256, 0, stream>>>((unsigned*)agg5, 0u, N_NODES * GAT_DIM);
    k_scatter_add_bf16<GAT_DIM><<<cdiv((long long)N_EDGES * (GAT_DIM / 4), 256), 256, 0, stream>>>(h1bf, src, dst, agg5);
    k_wmma_gemm<true, true, true, true, false><<<gblocks, 128, 0, stream>>>(
        agg5, h1bf, wW5r, wW5o, b5, h2, nullptr, N_NODES, GAT_DIM, HID);

    // ---- pool + MLP head ----
    k_fill_u32<<<cdiv(NUM_GRAPHS * HID, 256), 256, 0, stream>>>((unsigned*)g, 0u, NUM_GRAPHS * HID);
    k_pool<<<cdiv((long long)N_NODES * HID, 256), 256, 0, stream>>>(h2, batch, g);
    k_fc1<<<NUM_GRAPHS, HID, 0, stream>>>(g, W_fc1, b_fc1, t1);
    k_fc2<<<1, NUM_GRAPHS * NUM_CLASSES, 0, stream>>>(t1, W_fc2, b_fc2, (float*)d_out);

    (void)in_sizes; (void)n_in; (void)out_size; (void)ws_size;
}
